// BinaryTemporalBlock_77249281786110
// MI455X (gfx1250) — compile-verified, matrix operand order checked by reading, not run on hardware
//
#include <hip/hip_runtime.h>
#include <cstdint>
#include <cstddef>

// ---------------------------------------------------------------------------
// BinaryTemporalBlock (Conv-TasNet style) for gfx1250 / MI455X, wave32 + WMMA.
//   B=8, C=128, H=512, SC=128, T=4000, K=3, dilation=4, pad=4.
// Pipeline:
//   1. binarize w1 / (w2|w_skip stacked) / w_dw  -> bf16 weights in ws (w_dw kept fp32)
//   2. GEMM1 (WMMA bf16): h = W1b @ x + b1  (bf16 h in ws) + gLN1 partials
//   3. reduce -> mean/invstd (gLN1)
//   4. norm1+PReLU1 + depthwise dilated conv -> g (bf16 in ws) + gLN2 partials
//   5. reduce -> mean/invstd (gLN2)
//   6. GEMM2 (WMMA bf16): stacked [w2;w_skip](256x512) @ prelu(gln2(g));
//      rows 0..127 -> out = x + . + b2 ; rows 128..255 -> skip = . + b_skip
// ---------------------------------------------------------------------------

#define B_   8
#define C_   128
#define H_   512
#define T_   4000
#define TN   32           // time-tile (4000 % 32 == 0 -> no guards)
#define NT_TILES 125      // 4000 / 32
#define XPITCH (C_ + 8)   // LDS pitch (bf16 elems), multiple of 8 -> 16B rows
#define GPITCH (H_ + 8)   // 520, multiple of 8

typedef __attribute__((ext_vector_type(16))) __bf16 bf16x16;
typedef __attribute__((ext_vector_type(8)))  __bf16 bf16x8;
typedef __attribute__((ext_vector_type(8)))  float  f32x8;

// Load one 16-bf16 WMMA fragment: two contiguous 16B octets at p and p+16
// (matches the 16-bit A/B layout: per-lane K = kbase + {0..7, 16..23}).
__device__ __forceinline__ bf16x16 load_frag16(const __bf16* p) {
  bf16x8 lo = *(const bf16x8*)p;
  bf16x8 hi = *(const bf16x8*)(p + 16);
  bf16x16 r;
#pragma unroll
  for (int i = 0; i < 8; ++i) { r[i] = lo[i]; r[i + 8] = hi[i]; }
  return r;
}

// ------------------------- weight binarization -----------------------------

__global__ __launch_bounds__(128) void binarize_w1_kernel(
    const float* __restrict__ w1, __bf16* __restrict__ W1b) {
  const int o = blockIdx.x, tid = threadIdx.x;          // 512 blocks x 128 thr
  const float w = w1[o * C_ + tid];
  __shared__ float red[128];
  red[tid] = fabsf(w);
  __syncthreads();
  for (int off = 64; off; off >>= 1) {
    if (tid < off) red[tid] += red[tid + off];
    __syncthreads();
  }
  const float alpha = red[0] * (1.0f / C_);
  const float sg = (w > 0.f) ? 1.f : ((w < 0.f) ? -1.f : 0.f);
  W1b[o * C_ + tid] = (__bf16)(sg * alpha);
}

__global__ __launch_bounds__(256) void binarize_wo_kernel(
    const float* __restrict__ w2, const float* __restrict__ wsk,
    __bf16* __restrict__ Wob) {
  const int o = blockIdx.x, tid = threadIdx.x;          // 256 blocks x 256 thr
  const float* src = (o < 128) ? (w2 + o * H_) : (wsk + (o - 128) * H_);
  const float wa = src[tid], wb = src[tid + 256];
  __shared__ float red[256];
  red[tid] = fabsf(wa) + fabsf(wb);
  __syncthreads();
  for (int off = 128; off; off >>= 1) {
    if (tid < off) red[tid] += red[tid + off];
    __syncthreads();
  }
  const float alpha = red[0] * (1.0f / H_);
  const float sa = (wa > 0.f) ? 1.f : ((wa < 0.f) ? -1.f : 0.f);
  const float sb = (wb > 0.f) ? 1.f : ((wb < 0.f) ? -1.f : 0.f);
  Wob[o * H_ + tid]       = (__bf16)(sa * alpha);
  Wob[o * H_ + tid + 256] = (__bf16)(sb * alpha);
}

__global__ __launch_bounds__(256) void binarize_dw_kernel(
    const float* __restrict__ wdw, float* __restrict__ Wdwb) {
  const int h = blockIdx.x * 256 + threadIdx.x;         // 2 blocks x 256
  if (h >= H_) return;
  const float w0 = wdw[h * 3 + 0], w1 = wdw[h * 3 + 1], w2 = wdw[h * 3 + 2];
  const float alpha = (fabsf(w0) + fabsf(w1) + fabsf(w2)) * (1.0f / 3.0f);
  Wdwb[h * 3 + 0] = ((w0 > 0.f) ? alpha : ((w0 < 0.f) ? -alpha : 0.f));
  Wdwb[h * 3 + 1] = ((w1 > 0.f) ? alpha : ((w1 < 0.f) ? -alpha : 0.f));
  Wdwb[h * 3 + 2] = ((w2 > 0.f) ? alpha : ((w2 < 0.f) ? -alpha : 0.f));
}

// ------------------- GEMM1: h = W1b(512x128) @ x + b1 ----------------------
// grid (125, 8) x 256 threads (8 waves). Each wave: 4 M-tiles x 2 N-tiles.

__global__ __launch_bounds__(256) void gemm1_kernel(
    const float* __restrict__ x, const __bf16* __restrict__ W1b,
    const float* __restrict__ b1, __bf16* __restrict__ hb,
    float* __restrict__ partials) {
  const int tx = blockIdx.x, b = blockIdx.y;
  const int t0 = tx * TN;
  const int tid = threadIdx.x;
  const int lane = tid & 31, wave = tid >> 5;
  const int l15 = lane & 15;
  const int kbase = (lane >> 4) * 8;

  __shared__ __align__(16) __bf16 xs[TN * XPITCH];

  // Stage X tile [C=128][TN=32] into LDS, time-major [t][c] (coalesced reads).
#pragma unroll
  for (int i = 0; i < 16; ++i) {
    const int e = tid + i * 256;          // 0..4095
    const int c = e >> 5, t = e & 31;
    xs[t * XPITCH + c] = (__bf16)x[((size_t)b * C_ + c) * T_ + t0 + t];
  }

  const int m0 = wave * 64;
  // Prefetch our weight rows into GL2 (global_prefetch_b8).
  __builtin_prefetch(&W1b[(size_t)(m0 + (lane << 1)) * C_], 0, 3);
  __syncthreads();

  f32x8 acc[4][2];
#pragma unroll
  for (int mt = 0; mt < 4; ++mt)
#pragma unroll
    for (int nt = 0; nt < 2; ++nt) acc[mt][nt] = (f32x8)(0.0f);

#pragma unroll
  for (int k0 = 0; k0 < C_; k0 += 32) {
    bf16x16 bf[2];
#pragma unroll
    for (int nt = 0; nt < 2; ++nt)
      bf[nt] = load_frag16(&xs[(nt * 16 + l15) * XPITCH + k0 + kbase]);
#pragma unroll
    for (int mt = 0; mt < 4; ++mt) {
      const bf16x16 af =
          load_frag16(&W1b[(size_t)(m0 + mt * 16 + l15) * C_ + k0 + kbase]);
#pragma unroll
      for (int nt = 0; nt < 2; ++nt)
        acc[mt][nt] = __builtin_amdgcn_wmma_f32_16x16x32_bf16(
            false, af, false, bf[nt], (short)0, acc[mt][nt], false, false);
    }
  }

  // Store h (bf16) + accumulate gLN1 partial sums over this tile.
  const int mo = (lane >> 4) * 8;
  float s = 0.f, q = 0.f;
#pragma unroll
  for (int mt = 0; mt < 4; ++mt)
#pragma unroll
    for (int nt = 0; nt < 2; ++nt)
#pragma unroll
      for (int r = 0; r < 8; ++r) {
        const int row = m0 + mt * 16 + mo + r;
        const int t = t0 + nt * 16 + l15;
        const float v = acc[mt][nt][r] + b1[row];
        hb[((size_t)b * H_ + row) * T_ + t] = (__bf16)v;
        s += v;
        q += v * v;
      }
  for (int off = 16; off; off >>= 1) {
    s += __shfl_down(s, off, 32);
    q += __shfl_down(q, off, 32);
  }
  __shared__ float rs[8], rq[8];
  if (lane == 0) { rs[wave] = s; rq[wave] = q; }
  __syncthreads();
  if (tid == 0) {
    float S = 0.f, Q = 0.f;
    for (int w = 0; w < 8; ++w) { S += rs[w]; Q += rq[w]; }
    partials[((size_t)b * NT_TILES + tx) * 2 + 0] = S;
    partials[((size_t)b * NT_TILES + tx) * 2 + 1] = Q;
  }
}

// -------------------- stats finalize (shared by both gLNs) -----------------

__global__ __launch_bounds__(256) void reduce_stats_kernel(
    const float* __restrict__ partials, int n, float* __restrict__ stats,
    float invN) {
  const int b = blockIdx.x, tid = threadIdx.x;
  float s = 0.f, q = 0.f;
  for (int i = tid; i < n; i += 256) {
    s += partials[((size_t)b * n + i) * 2 + 0];
    q += partials[((size_t)b * n + i) * 2 + 1];
  }
  __shared__ float rs[256], rq[256];
  rs[tid] = s; rq[tid] = q;
  __syncthreads();
  for (int off = 128; off; off >>= 1) {
    if (tid < off) { rs[tid] += rs[tid + off]; rq[tid] += rq[tid + off]; }
    __syncthreads();
  }
  if (tid == 0) {
    const float mean = rs[0] * invN;
    const float var = rq[0] * invN - mean * mean;
    stats[b * 2 + 0] = mean;
    stats[b * 2 + 1] = rsqrtf(var + 1e-8f);
  }
}

// -------- norm1 + PReLU1 + depthwise dilated conv (K=3, dil=4) -------------
// grid (16, 512, 8) x 256: one (b, ch, 256-wide t chunk) per block.

__global__ __launch_bounds__(256) void dwconv_kernel(
    const __bf16* __restrict__ hb, const float* __restrict__ Wdwb,
    const float* __restrict__ g1v, const float* __restrict__ be1v,
    const float* __restrict__ p1p, const float* __restrict__ stats1,
    __bf16* __restrict__ g, float* __restrict__ partials) {
  const int chunk = blockIdx.x, ch = blockIdx.y, b = blockIdx.z;
  const int tid = threadIdx.x;
  const int t = chunk * 256 + tid;
  const float m = stats1[b * 2 + 0], inv = stats1[b * 2 + 1];
  const float ga = g1v[ch], be = be1v[ch], p1 = *p1p;
  const __bf16* hrow = hb + ((size_t)b * H_ + ch) * T_;
  float y = 0.f;
  if (t < T_) {
    const float w0 = Wdwb[ch * 3 + 0];
    const float w1 = Wdwb[ch * 3 + 1];
    const float w2 = Wdwb[ch * 3 + 2];
    float u0 = 0.f, u1, u2 = 0.f;
    if (t - 4 >= 0) {
      const float v = ((float)hrow[t - 4] - m) * inv * ga + be;
      u0 = (v >= 0.f) ? v : p1 * v;
    }
    {
      const float v = ((float)hrow[t] - m) * inv * ga + be;
      u1 = (v >= 0.f) ? v : p1 * v;
    }
    if (t + 4 < T_) {
      const float v = ((float)hrow[t + 4] - m) * inv * ga + be;
      u2 = (v >= 0.f) ? v : p1 * v;
    }
    y = w0 * u0 + w1 * u1 + w2 * u2;
    g[((size_t)b * H_ + ch) * T_ + t] = (__bf16)y;
  }
  float s = y, q = y * y;
  for (int off = 16; off; off >>= 1) {
    s += __shfl_down(s, off, 32);
    q += __shfl_down(q, off, 32);
  }
  __shared__ float rs[8], rq[8];
  const int lane = tid & 31, wave = tid >> 5;
  if (lane == 0) { rs[wave] = s; rq[wave] = q; }
  __syncthreads();
  if (tid == 0) {
    float S = 0.f, Q = 0.f;
    for (int w = 0; w < 8; ++w) { S += rs[w]; Q += rq[w]; }
    const size_t p = (size_t)b * (512 * 16) + ch * 16 + chunk;
    partials[p * 2 + 0] = S;
    partials[p * 2 + 1] = Q;
  }
}

// ------ GEMM2: [out;skip](256x512) @ prelu(gln2(g)) , + residual + bias ----
// grid (125, 8) x 256 threads. Each wave: 2 M-tiles x 2 N-tiles.

__global__ __launch_bounds__(256) void gemm2_kernel(
    const __bf16* __restrict__ g, const __bf16* __restrict__ Wob,
    const float* __restrict__ g2v, const float* __restrict__ be2v,
    const float* __restrict__ p2p, const float* __restrict__ stats2,
    const float* __restrict__ x, const float* __restrict__ b2,
    const float* __restrict__ bsk, float* __restrict__ out) {
  const int tx = blockIdx.x, b = blockIdx.y;
  const int t0 = tx * TN;
  const int tid = threadIdx.x;
  const int lane = tid & 31, wave = tid >> 5;
  const int l15 = lane & 15;
  const int kbase = (lane >> 4) * 8;

  __shared__ __align__(16) __bf16 gs[TN * GPITCH];   // 32 x 520 bf16 = 33,280B

  const float m2 = stats2[b * 2 + 0], inv2 = stats2[b * 2 + 1];
  const float p2 = *p2p;

  // Stage normalized+PReLU'd G tile [H=512][TN=32] into LDS, time-major.
#pragma unroll
  for (int i = 0; i < 64; ++i) {
    const int e = tid + i * 256;          // 0..16383
    const int ch = e >> 5, t = e & 31;
    const float raw = (float)g[((size_t)b * H_ + ch) * T_ + t0 + t];
    float v = (raw - m2) * inv2 * g2v[ch] + be2v[ch];
    v = (v >= 0.f) ? v : p2 * v;
    gs[t * GPITCH + ch] = (__bf16)v;
  }

  const int m0 = wave * 32;
  __builtin_prefetch(&Wob[(size_t)(m0 + lane) * H_], 0, 3);
  __syncthreads();

  f32x8 acc[2][2];
#pragma unroll
  for (int mt = 0; mt < 2; ++mt)
#pragma unroll
    for (int nt = 0; nt < 2; ++nt) acc[mt][nt] = (f32x8)(0.0f);

#pragma unroll 4
  for (int kk = 0; kk < 16; ++kk) {
    const int k0 = kk * 32;
    bf16x16 bf[2];
#pragma unroll
    for (int nt = 0; nt < 2; ++nt)
      bf[nt] = load_frag16(&gs[(nt * 16 + l15) * GPITCH + k0 + kbase]);
#pragma unroll
    for (int mt = 0; mt < 2; ++mt) {
      const bf16x16 af =
          load_frag16(&Wob[(size_t)(m0 + mt * 16 + l15) * H_ + k0 + kbase]);
#pragma unroll
      for (int nt = 0; nt < 2; ++nt)
        acc[mt][nt] = __builtin_amdgcn_wmma_f32_16x16x32_bf16(
            false, af, false, bf[nt], (short)0, acc[mt][nt], false, false);
    }
  }

  // rows 0..127 -> out = x + acc + b2 ; rows 128..255 -> skip = acc + b_skip.
  const int mo = (lane >> 4) * 8;
  const size_t SKIP_OFF = (size_t)B_ * C_ * T_;   // out|skip concatenated
#pragma unroll
  for (int mt = 0; mt < 2; ++mt)
#pragma unroll
    for (int nt = 0; nt < 2; ++nt)
#pragma unroll
      for (int r = 0; r < 8; ++r) {
        const int row = m0 + mt * 16 + mo + r;
        const int t = t0 + nt * 16 + l15;
        float v = acc[mt][nt][r];
        if (row < 128) {
          const size_t idx = ((size_t)b * C_ + row) * T_ + t;
          out[idx] = v + b2[row] + x[idx];
        } else {
          const int c = row - 128;
          out[SKIP_OFF + ((size_t)b * C_ + c) * T_ + t] = v + bsk[c];
        }
      }
}

// ---------------------------------------------------------------------------

extern "C" void kernel_launch(void* const* d_in, const int* in_sizes, int n_in,
                              void* d_out, int out_size, void* d_ws,
                              size_t ws_size, hipStream_t stream) {
  const float* x     = (const float*)d_in[0];
  const float* w1    = (const float*)d_in[1];
  const float* b1    = (const float*)d_in[2];
  const float* g1v   = (const float*)d_in[3];
  const float* be1v  = (const float*)d_in[4];
  const float* p1p   = (const float*)d_in[5];
  const float* wdw   = (const float*)d_in[6];
  const float* g2v   = (const float*)d_in[7];
  const float* be2v  = (const float*)d_in[8];
  const float* p2p   = (const float*)d_in[9];
  const float* w2    = (const float*)d_in[10];
  const float* b2    = (const float*)d_in[11];
  const float* wsk   = (const float*)d_in[12];
  const float* bsk   = (const float*)d_in[13];
  float* out = (float*)d_out;

  uint8_t* ws = (uint8_t*)d_ws;
  size_t off = 0;
  auto carve = [&](size_t bytes) -> void* {
    void* p = ws + off;
    off = (off + bytes + 255) & ~(size_t)255;
    return p;
  };
  __bf16* HB  = (__bf16*)carve((size_t)B_ * H_ * T_ * 2);   // conv1 output
  __bf16* G   = (__bf16*)carve((size_t)B_ * H_ * T_ * 2);   // dwconv output
  __bf16* W1B = (__bf16*)carve((size_t)H_ * C_ * 2);
  __bf16* WOB = (__bf16*)carve((size_t)256 * H_ * 2);
  float*  WDW = (float*)carve((size_t)H_ * 3 * 4);
  float*  P1  = (float*)carve((size_t)B_ * NT_TILES * 2 * 4);
  float*  P2  = (float*)carve((size_t)B_ * 512 * 16 * 2 * 4);
  float*  S1  = (float*)carve(64);
  float*  S2  = (float*)carve(64);

  const float invN = 1.0f / ((float)H_ * (float)T_);

  binarize_w1_kernel<<<H_, 128, 0, stream>>>(w1, W1B);
  binarize_wo_kernel<<<256, 256, 0, stream>>>(w2, wsk, WOB);
  binarize_dw_kernel<<<2, 256, 0, stream>>>(wdw, WDW);

  gemm1_kernel<<<dim3(NT_TILES, B_), 256, 0, stream>>>(x, W1B, b1, HB, P1);
  reduce_stats_kernel<<<B_, 256, 0, stream>>>(P1, NT_TILES, S1, invN);

  dwconv_kernel<<<dim3(16, 512, B_), 256, 0, stream>>>(HB, WDW, g1v, be1v,
                                                       p1p, S1, G, P2);
  reduce_stats_kernel<<<B_, 256, 0, stream>>>(P2, 512 * 16, S2, invN);

  gemm2_kernel<<<dim3(NT_TILES, B_), 256, 0, stream>>>(G, WOB, g2v, be2v, p2p,
                                                       S2, x, b2, bsk, out);
}
